// EncoderLayer_53437983097250
// MI455X (gfx1250) — matrix operand order repaired and checked by reading, not compiled, not run on hardware
//
#include <hip/hip_runtime.h>
#include <cstddef>
#include <cstdint>

// ---------------------------------------------------------------------------
// Types for CDNA5 WMMA (wave32): V_WMMA_F32_16X16X32_F16
// ---------------------------------------------------------------------------
typedef _Float16 h16;
typedef __attribute__((ext_vector_type(16))) _Float16 v16h;
typedef __attribute__((ext_vector_type(8)))  _Float16 v8h;
typedef __attribute__((ext_vector_type(8)))  float    v8f;
typedef __attribute__((ext_vector_type(4)))  unsigned int u32x4;
typedef __attribute__((ext_vector_type(8)))  int          i32x8;
typedef __attribute__((ext_vector_type(4)))  int          i32x4;

__device__ __forceinline__ v8f wmma16x16x32(v16h a, v16h b, v8f c) {
    return __builtin_amdgcn_wmma_f32_16x16x32_f16(
        /*neg_a=*/false, a, /*neg_b=*/false, b,
        /*c_mod=*/(short)0, c, /*reuse_a=*/false, /*reuse_b=*/false);
}

// Problem constants
#define BATCH 2
#define SEQ   2048
#define DMODEL 1024
#define NHEAD 16
#define DHEAD 64
#define HIDDEN 4096
#define NTOK  (BATCH * SEQ)          // 4096

// ---------------------------------------------------------------------------
// Tensor Data Mover: 2D f16 tile load Global -> LDS.
// This toolchain exposes the 6-arg builtin:
//   (uint32x4 g0, int32x8 g1, int32x4 g2, int32x4 g3, int32x8 extra, i32 cpol)
// D# group0: [1:0]=count=1, [63:32]=lds_addr, [120:64]=global_addr,
//            [127:126]=type=2.  group1: data_size=1(2B), tensor dims/strides,
//            tile dims.  Groups 2/3 zero (2D tensor).
// ---------------------------------------------------------------------------
__device__ __forceinline__ void tdm_load_2d_f16(const h16* gsrc, unsigned lds_addr,
                                                int tileW, int tileH,
                                                int tensorW, int tensorH) {
    unsigned long long ga = (unsigned long long)(uintptr_t)gsrc;
    u32x4 g0;
    g0[0] = 1u;                                          // count=1 (valid), user mode
    g0[1] = lds_addr;                                    // LDS byte address
    g0[2] = (unsigned)(ga & 0xffffffffu);                // global_addr[31:0]
    g0[3] = (unsigned)((ga >> 32) & 0x01ffffffu)         // global_addr[56:32]
          | (2u << 30);                                  // type=2 ("image")
    i32x8 g1;
    g1[0] = (int)(1u << 16);                             // data_size=1 -> 2 bytes
    g1[1] = (int)(((unsigned)tensorW & 0xffffu) << 16);  // tensor_dim0[15:0]
    g1[2] = (int)((((unsigned)tensorW >> 16) & 0xffffu)
          | (((unsigned)tensorH & 0xffffu) << 16));      // dim0 hi | dim1 lo
    g1[3] = (int)((((unsigned)tensorH >> 16) & 0xffffu)
          | (((unsigned)tileW & 0xffffu) << 16));        // dim1 hi | tile_dim0
    g1[4] = (int)((unsigned)tileH & 0xffffu);            // tile_dim1 (tile_dim2=0)
    g1[5] = (int)(unsigned)tensorW;                      // tensor_dim0_stride[31:0]
    g1[6] = 0;                                           // stride hi | dim1_stride lo
    g1[7] = 0;
    i32x4 z4 = {0, 0, 0, 0};
    i32x8 z8 = {0, 0, 0, 0, 0, 0, 0, 0};
    __builtin_amdgcn_tensor_load_to_lds(g0, g1, z4, z4, z8, 0);
}

// ---------------------------------------------------------------------------
// f32 -> f16 convert (n divisible by 4)
// ---------------------------------------------------------------------------
__global__ __launch_bounds__(256)
void k_f2h(const float* __restrict__ src, h16* __restrict__ dst, int n) {
    int i = (blockIdx.x * 256 + threadIdx.x) * 4;
    if (i < n) {
        float4 v = *(const float4*)(src + i);
        dst[i + 0] = (h16)v.x;
        dst[i + 1] = (h16)v.y;
        dst[i + 2] = (h16)v.z;
        dst[i + 3] = (h16)v.w;
    }
}

// ---------------------------------------------------------------------------
// WMMA GEMM: C[M,N] = A[M,K] @ B[K,N], A/B f16 row-major.
// Workgroup = 256 threads (8 waves), tile 128(M) x 128(N), K-step 32.
// Waves in a 4x2 grid: each wave owns a 32x64 sub-tile = 2x4 WMMA tiles,
// so 8 WMMAs amortize 2 A-frag + 4 B-frag LDS reads per step.
// A tile is DMA'd by the Tensor Data Mover (triple-buffered, 1 step ahead);
// B tile is staged + transposed by the threads (double-buffered).
// ---------------------------------------------------------------------------
enum {
    EPI_F16 = 0,       // plain f16 store
    EPI_QK_HEAD = 1,   // store f16 as [b,h,s,d]
    EPI_V_HEADT = 2,   // store f16 as [b,h,d,s] (transposed per head)
    EPI_WO_RES = 3,    // f32 store: val + residual  (pre-LN1)
    EPI_FC1_GELU = 4,  // f16 store: gelu(val + bias)
    EPI_FC2_OUT = 5    // f32 store: xres + gamma*(val + bias)
};

__global__ __launch_bounds__(256)
void k_gemm(const h16* __restrict__ A, const h16* __restrict__ B,
            int M, int N, int K, int epi,
            h16* __restrict__ outh, float* __restrict__ outf,
            const float* __restrict__ bias, const float* __restrict__ res,
            const float* __restrict__ gamma)
{
    __shared__ h16 sA[3][128 * 32];   // triple-buffered TDM destination
    __shared__ h16 sBT[2][128 * 48];  // B transposed [n][k], padded to 48

    const int tid  = threadIdx.x;
    const int lane = tid & 31;
    const int wave = tid >> 5;
    const int wm   = wave >> 1;       // 0..3 : 32-row strip
    const int wn   = wave & 1;        // 0..1 : 64-col strip
    const int rowBase = blockIdx.y * 128;
    const int colBase = blockIdx.x * 128;
    const int l15 = lane & 15;
    const int lhi = lane >> 4;        // 0 or 1
    const int steps = K >> 5;

    v8f acc[2][4];
    #pragma unroll
    for (int a = 0; a < 2; ++a)
        #pragma unroll
        for (int n = 0; n < 4; ++n) acc[a][n] = v8f{};

    // Prologue: TDM fetch of A tile for step 0 (wave 0 issues; EXEC-independent)
    if (tid < 32) {
        tdm_load_2d_f16(&A[(size_t)rowBase * K],
                        (unsigned)(uintptr_t)&sA[0][0], 32, 128, K, M);
    }

    const int bk = tid >> 3;          // 0..31 : k within step
    const int bn = (tid & 7) * 16;    // 0..112: n segment

    for (int i = 0; i < steps; ++i) {
        const int k0 = i << 5;
        // ---- stage+transpose B(i): one 32B vector load, 16 ds_store_b16 ----
        {
            const h16* src = &B[(size_t)(k0 + bk) * N + colBase + bn];
            v16h bv = *(const v16h*)src;
            if (i + 1 < steps)
                __builtin_prefetch(src + (size_t)32 * N, 0, 1);  // next B tile
            h16* dst = &sBT[i & 1][0];
            #pragma unroll
            for (int e = 0; e < 16; ++e) dst[(bn + e) * 48 + bk] = bv[e];
        }
        // ---- TDM: issue A(i+1), then wait so A(i) is resident ----
        if (tid < 32) {
            if (i + 1 < steps) {
                tdm_load_2d_f16(&A[(size_t)rowBase * K + (k0 + 32)],
                                (unsigned)(uintptr_t)&sA[(i + 1) % 3][0],
                                32, 128, K, M);
                __builtin_amdgcn_s_wait_tensorcnt(1);
            } else {
                __builtin_amdgcn_s_wait_tensorcnt(0);
            }
        }
        __syncthreads();   // A(i) + B(i) visible to all waves

        const h16* sa = &sA[i % 3][0];
        const h16* sb = &sBT[i & 1][0];

        v16h af[2];
        #pragma unroll
        for (int a = 0; a < 2; ++a) {
            const h16* p = &sa[(wm * 32 + a * 16 + l15) * 32 + (lhi << 4)];
            v8h lo = *(const v8h*)p;
            v8h hi = *(const v8h*)(p + 8);
            #pragma unroll
            for (int h = 0; h < 8; ++h) { af[a][h] = lo[h]; af[a][h + 8] = hi[h]; }
        }
        #pragma unroll
        for (int n = 0; n < 4; ++n) {
            const v16h bf =
                *(const v16h*)(&sb[(wn * 64 + n * 16 + l15) * 48 + (lhi << 4)]);
            acc[0][n] = wmma16x16x32(af[0], bf, acc[0][n]);
            acc[1][n] = wmma16x16x32(af[1], bf, acc[1][n]);
        }
        // No trailing barrier: next iteration writes the *other* sBT buffer and
        // a different sA buffer; a buffer is rewritten only >=1 barrier after
        // its last reader (triple-buffered sA, double-buffered sBT).
    }

    // ---- epilogue ----
    #pragma unroll
    for (int a = 0; a < 2; ++a) {
        const int mrow0 = rowBase + wm * 32 + a * 16 + (lhi << 3);
        #pragma unroll
        for (int n = 0; n < 4; ++n) {
            const int col = colBase + wn * 64 + n * 16 + l15;
            #pragma unroll
            for (int r = 0; r < 8; ++r) {
                const int row = mrow0 + r;
                const float v = acc[a][n][r];
                switch (epi) {
                    case EPI_F16:
                        outh[(size_t)row * N + col] = (h16)v;
                        break;
                    case EPI_QK_HEAD: {
                        int b = row >> 11, s = row & 2047;
                        int hh = col >> 6, d = col & 63;
                        outh[((((size_t)b * NHEAD + hh) * SEQ + s) << 6) + d] = (h16)v;
                    } break;
                    case EPI_V_HEADT: {
                        int b = row >> 11, s = row & 2047;
                        int hh = col >> 6, d = col & 63;
                        outh[(((size_t)b * NHEAD + hh) * DHEAD + d) * SEQ + s] = (h16)v;
                    } break;
                    case EPI_WO_RES: {
                        size_t idx = (size_t)row * N + col;
                        outf[idx] = v + res[idx];
                    } break;
                    case EPI_FC1_GELU: {
                        float x = v + bias[col];
                        float g = 0.5f * x * (1.0f + erff(x * 0.70710678118654752f));
                        outh[(size_t)row * N + col] = (h16)g;
                    } break;
                    case EPI_FC2_OUT: {
                        size_t idx = (size_t)row * N + col;
                        float x = v + bias[col];
                        outf[idx] = res[idx] + gamma[col] * x;
                    } break;
                }
            }
        }
    }
}

// ---------------------------------------------------------------------------
// Attention kernel. One workgroup (128 threads = 4 waves) per
// (b, h, 16-query block). Full 16x2048 score slab lives in LDS (128 KB),
// so the 536 MB attn tensor is written to HBM exactly once and the P@V
// GEMM reads probabilities straight from LDS.
//   Q: [bh][s][64] f16   K: [bh][s][64] f16   VT: [bh][64][s] f16
//   attn_out: [bh][q][k] f32   ctx: [tok][h*64+d] f16
// ---------------------------------------------------------------------------
__global__ __launch_bounds__(128)
void k_attn(const h16* __restrict__ Qh, const h16* __restrict__ Kh,
            const h16* __restrict__ VTh, const unsigned char* __restrict__ mask,
            float* __restrict__ attn_out, h16* __restrict__ ctxh)
{
    __shared__ float sc[16 * SEQ];   // 128 KB score/prob slab
    __shared__ h16   qt[16 * DHEAD]; // 2 KB Q block

    const int blk  = blockIdx.x;     // 0..4095
    const int qb   = blk & 127;      // query block (16 queries each)
    const int bh   = blk >> 7;       // b*16 + h
    const int b    = bh >> 4;
    const int tid  = threadIdx.x;
    const int lane = tid & 31;
    const int wave = tid >> 5;       // 0..3
    const int l15  = lane & 15;
    const int lhi  = lane >> 4;

    const size_t headOff = (size_t)bh * SEQ * DHEAD;
    const h16* Qbase  = Qh  + headOff + (size_t)qb * 16 * DHEAD;
    const h16* Kbase  = Kh  + headOff;
    const h16* VTbase = VTh + headOff;

    // ---- load Q block (16x64 halves) ----
    *(v8h*)(&qt[tid * 8]) = *(const v8h*)(Qbase + tid * 8);
    __syncthreads();

    // ---- build the two Q A-fragments (K=0..31 and K=32..63) ----
    v16h aq[2];
    #pragma unroll
    for (int kk = 0; kk < 2; ++kk) {
        const h16* p = &qt[l15 * DHEAD + kk * 32 + (lhi << 4)];
        v8h lo = *(const v8h*)p;
        v8h hi = *(const v8h*)(p + 8);
        #pragma unroll
        for (int h = 0; h < 8; ++h) { aq[kk][h] = lo[h]; aq[kk][h + 8] = hi[h]; }
    }

    // ---- score phase: S = Q K^T, tiles strided over the 4 waves ----
    for (int kt = wave; kt < 128; kt += 4) {
        v8f c = v8f{};
        #pragma unroll
        for (int kk = 0; kk < 2; ++kk) {
            // B-frag = K^T chunk: one contiguous 32B load per lane from K row
            const v16h bf = *(const v16h*)(
                Kbase + (size_t)(kt * 16 + l15) * DHEAD + kk * 32 + (lhi << 4));
            c = wmma16x16x32(aq[kk], bf, c);
        }
        const int kcol = kt * 16 + l15;
        const int mrow = lhi << 3;
        #pragma unroll
        for (int r = 0; r < 8; ++r) sc[(mrow + r) * SEQ + kcol] = c[r];
    }
    __syncthreads();

    // ---- softmax + mask in LDS; 8 threads per row, 256 cols each ----
    {
        const int row = tid >> 3;     // 0..15
        const int seg = tid & 7;      // 0..7
        const float scale = 0.125f;   // 1/sqrt(64)
        float4* srow4 = (float4*)&sc[row * SEQ + seg * 256];
        const uchar4* m4 = (const uchar4*)(
            mask + ((size_t)b * SEQ + (qb * 16 + row)) * SEQ + seg * 256);

        float mx = -3.4e38f;
        for (int i = 0; i < 64; ++i) {
            float4 sv = srow4[i];
            uchar4 mv = m4[i];
            sv.x = mv.x ? -1e9f : sv.x * scale;
            sv.y = mv.y ? -1e9f : sv.y * scale;
            sv.z = mv.z ? -1e9f : sv.z * scale;
            sv.w = mv.w ? -1e9f : sv.w * scale;
            srow4[i] = sv;
            mx = fmaxf(mx, fmaxf(fmaxf(sv.x, sv.y), fmaxf(sv.z, sv.w)));
        }
        #pragma unroll
        for (int off = 1; off < 8; off <<= 1) mx = fmaxf(mx, __shfl_xor(mx, off, 32));

        float sum = 0.f;
        for (int i = 0; i < 64; ++i) {
            float4 sv = srow4[i];
            sv.x = __expf(sv.x - mx);
            sv.y = __expf(sv.y - mx);
            sv.z = __expf(sv.z - mx);
            sv.w = __expf(sv.w - mx);
            srow4[i] = sv;
            sum += sv.x + sv.y + sv.z + sv.w;
        }
        #pragma unroll
        for (int off = 1; off < 8; off <<= 1) sum += __shfl_xor(sum, off, 32);
        const float inv = 1.0f / sum;

        float4* aout4 = (float4*)(
            attn_out + ((size_t)bh * SEQ + qb * 16 + row) * SEQ + seg * 256);
        for (int i = 0; i < 64; ++i) {
            float4 sv = srow4[i];
            sv.x *= inv; sv.y *= inv; sv.z *= inv; sv.w *= inv;
            srow4[i] = sv;        // keep normalized probs in LDS for P@V
            aout4[i] = sv;        // single HBM write of attn tensor
        }
    }
    __syncthreads();

    // ---- ctx = P @ V : each wave owns one 16-wide d-column tile ----
    const int n = wave;  // d tile 0..3
    v8f c = v8f{};
    for (int it = 0; it < 64; ++it) {     // K = 2048 in steps of 32
        // A-frag from LDS probs (two contiguous 8-float chunks -> f16)
        const float* pp = &sc[l15 * SEQ + it * 32 + (lhi << 4)];
        v16h af;
        #pragma unroll
        for (int h = 0; h < 8; ++h) {
            af[h]     = (h16)pp[h];
            af[h + 8] = (h16)pp[h + 8];
        }
        // B-frag from VT (contiguous 32B per lane)
        const v16h bf = *(const v16h*)(
            VTbase + (size_t)(n * 16 + l15) * SEQ + it * 32 + (lhi << 4));
        c = wmma16x16x32(af, bf, c);
    }
    {
        const int hh  = bh & 15;
        const int col = hh * DHEAD + n * 16 + l15;
        #pragma unroll
        for (int r = 0; r < 8; ++r) {
            const int s = qb * 16 + (lhi << 3) + r;
            ctxh[((size_t)(b * SEQ + s)) * DMODEL + col] = (h16)c[r];
        }
    }
}

// ---------------------------------------------------------------------------
// Fused LayerNorm pair: x = LN(pre) (no affine); y = LN(x)*w + b (f16).
// One workgroup (256 threads) per row of 1024.
// ---------------------------------------------------------------------------
__device__ __forceinline__ void block_reduce2(float& a, float& b2, float* red) {
    #pragma unroll
    for (int off = 16; off > 0; off >>= 1) {
        a  += __shfl_xor(a,  off, 32);
        b2 += __shfl_xor(b2, off, 32);
    }
    const int wv = threadIdx.x >> 5;
    if ((threadIdx.x & 31) == 0) { red[wv] = a; red[8 + wv] = b2; }
    __syncthreads();
    float ta = 0.f, tb = 0.f;
    #pragma unroll
    for (int i = 0; i < 8; ++i) { ta += red[i]; tb += red[8 + i]; }
    a = ta; b2 = tb;
    __syncthreads();
}

__global__ __launch_bounds__(256)
void k_ln(const float* __restrict__ pre, float* __restrict__ xout,
          h16* __restrict__ yout, const float* __restrict__ w,
          const float* __restrict__ bia)
{
    __shared__ float red[16];
    const size_t row = blockIdx.x;
    const float* p = pre + row * DMODEL;
    const int t = threadIdx.x;

    float v[4], s = 0.f, s2 = 0.f;
    #pragma unroll
    for (int i = 0; i < 4; ++i) {
        v[i] = p[t + i * 256];
        s += v[i]; s2 += v[i] * v[i];
    }
    block_reduce2(s, s2, red);
    const float mu   = s * (1.0f / DMODEL);
    const float var  = s2 * (1.0f / DMODEL) - mu * mu;
    const float rstd = rsqrtf(var + 1e-5f);

    float x[4]; s = 0.f; s2 = 0.f;
    #pragma unroll
    for (int i = 0; i < 4; ++i) {
        x[i] = (v[i] - mu) * rstd;
        xout[row * DMODEL + t + i * 256] = x[i];
        s += x[i]; s2 += x[i] * x[i];
    }
    block_reduce2(s, s2, red);
    const float mu2   = s * (1.0f / DMODEL);
    const float var2  = s2 * (1.0f / DMODEL) - mu2 * mu2;
    const float rstd2 = rsqrtf(var2 + 1e-5f);
    #pragma unroll
    for (int i = 0; i < 4; ++i) {
        const int c = t + i * 256;
        yout[row * DMODEL + c] = (h16)((x[i] - mu2) * rstd2 * w[c] + bia[c]);
    }
}

// ---------------------------------------------------------------------------
// Host-side launch
// ---------------------------------------------------------------------------
extern "C" void kernel_launch(void* const* d_in, const int* in_sizes, int n_in,
                              void* d_out, int out_size, void* d_ws, size_t ws_size,
                              hipStream_t stream) {
    (void)in_sizes; (void)n_in; (void)out_size; (void)ws_size;

    const float*         enc_in = (const float*)d_in[0];
    const unsigned char* mask   = (const unsigned char*)d_in[1];  // bool array
    const float* Wq   = (const float*)d_in[2];
    const float* Wk   = (const float*)d_in[3];
    const float* Wv   = (const float*)d_in[4];
    const float* Wo   = (const float*)d_in[5];
    const float* ln2w = (const float*)d_in[6];
    const float* ln2b = (const float*)d_in[7];
    const float* fc1w = (const float*)d_in[8];
    const float* fc1b = (const float*)d_in[9];
    const float* fc2w = (const float*)d_in[10];
    const float* fc2b = (const float*)d_in[11];
    const float* gam2 = (const float*)d_in[12];

    float* out_enc  = (float*)d_out;                                  // [4096,1024]
    float* out_attn = out_enc + (size_t)NTOK * DMODEL;                // [32,2048,2048]

    const size_t MB = 1u << 20;
    char* ws = (char*)d_ws;
    h16*   xh    = (h16*)(ws + 0 * MB);     // 8 MB  x in f16
    h16*   wqh   = (h16*)(ws + 8 * MB);     // 2 MB
    h16*   wkh   = (h16*)(ws + 10 * MB);
    h16*   wvh   = (h16*)(ws + 12 * MB);
    h16*   woh   = (h16*)(ws + 14 * MB);
    h16*   fc1wh = (h16*)(ws + 16 * MB);    // 8 MB
    h16*   fc2wh = (h16*)(ws + 24 * MB);    // 8 MB
    h16*   Qhp   = (h16*)(ws + 32 * MB);    // 8 MB [bh][s][64]
    h16*   Khp   = (h16*)(ws + 40 * MB);    // 8 MB [bh][s][64]
    h16*   VThp  = (h16*)(ws + 48 * MB);    // 8 MB [bh][64][s]
    h16*   ctxh  = (h16*)(ws + 56 * MB);    // 8 MB [tok][1024]
    float* pre   = (float*)(ws + 64 * MB);  // 16 MB  ctx@Wo + residual
    float* xbuf  = (float*)(ws + 80 * MB);  // 16 MB  LN1 output
    h16*   ybuf  = (h16*)(ws + 96 * MB);    // 8 MB   LN2 output f16
    h16*   hbuf  = (h16*)(ws + 32 * MB);    // 32 MB  gelu(fc1) — reuses Q/K/V/ctx

    // 1) f32 -> f16 conversions
    auto cv = [&](const float* s, h16* d, int n) {
        k_f2h<<<dim3(n / 1024), dim3(256), 0, stream>>>(s, d, n);
    };
    cv(enc_in, xh,    NTOK * DMODEL);
    cv(Wq,     wqh,   DMODEL * DMODEL);
    cv(Wk,     wkh,   DMODEL * DMODEL);
    cv(Wv,     wvh,   DMODEL * DMODEL);
    cv(Wo,     woh,   DMODEL * DMODEL);
    cv(fc1w,   fc1wh, DMODEL * HIDDEN);
    cv(fc2w,   fc2wh, HIDDEN * DMODEL);

    // 2) Q/K/V projections (head layouts baked in the epilogue)
    {
        dim3 g(DMODEL / 128, NTOK / 128), blk(256);
        k_gemm<<<g, blk, 0, stream>>>(xh, wqh, NTOK, DMODEL, DMODEL,
                                      EPI_QK_HEAD, Qhp, nullptr, nullptr, nullptr, nullptr);
        k_gemm<<<g, blk, 0, stream>>>(xh, wkh, NTOK, DMODEL, DMODEL,
                                      EPI_QK_HEAD, Khp, nullptr, nullptr, nullptr, nullptr);
        k_gemm<<<g, blk, 0, stream>>>(xh, wvh, NTOK, DMODEL, DMODEL,
                                      EPI_V_HEADT, VThp, nullptr, nullptr, nullptr, nullptr);
    }

    // 3) Attention: scores + mask + softmax (LDS-resident) + attn write + P@V
    k_attn<<<dim3(BATCH * NHEAD * (SEQ / 16)), dim3(128), 0, stream>>>(
        Qhp, Khp, VThp, mask, out_attn, ctxh);

    // 4) ctx @ Wo + residual -> pre-LN1
    k_gemm<<<dim3(DMODEL / 128, NTOK / 128), dim3(256), 0, stream>>>(
        ctxh, woh, NTOK, DMODEL, DMODEL, EPI_WO_RES,
        nullptr, pre, nullptr, enc_in, nullptr);

    // 5) fused LN1 (no affine) + LN2 (affine, f16 out)
    k_ln<<<dim3(NTOK), dim3(256), 0, stream>>>(pre, xbuf, ybuf, ln2w, ln2b);

    // 6) fc1 + exact GELU
    k_gemm<<<dim3(HIDDEN / 128, NTOK / 128), dim3(256), 0, stream>>>(
        ybuf, fc1wh, NTOK, HIDDEN, DMODEL, EPI_FC1_GELU,
        hbuf, nullptr, fc1b, nullptr, nullptr);

    // 7) fc2 + layer-scale residual -> enc_outputs
    k_gemm<<<dim3(DMODEL / 128, NTOK / 128), dim3(256), 0, stream>>>(
        hbuf, fc2wh, NTOK, DMODEL, HIDDEN, EPI_FC2_OUT,
        nullptr, out_enc, fc2b, xbuf, gam2);
}
